// Model_77086073028807
// MI455X (gfx1250) — matrix operand order repaired and checked by reading, not compiled, not run on hardware
//
#include <hip/hip_runtime.h>

#define N_NODES 100000
#define N_EDGES 1600000
#define F_IN    64
#define CH      128
#define CH3     384
#define STEPS   4
#define NEG_SLOPE 0.01f

typedef __attribute__((ext_vector_type(16))) __bf16 bf16x16;
typedef __attribute__((ext_vector_type(8)))  float  v8f;

union FragU { unsigned int u[8]; bf16x16 v; };

__device__ __forceinline__ unsigned short f2bf(float f) {
  unsigned int u = __float_as_uint(f);
  u += 0x7FFFu + ((u >> 16) & 1u);           // round-to-nearest-even
  return (unsigned short)(u >> 16);
}

// ---------------------------------------------------------------------------
// Pack a weight matrix into WMMA B-fragment layout (bf16), contiguous k order:
// B is 32x16 (KxN) per tile; lane<16 -> N=lane, K=kt*32+{0..15}; lane>=16 ->
// N=lane-16, K=kt*32+{16..31}; VGPR v holds pair (K=2v, K=2v+1).
// B(k,n) = transposed ? W[n*ld + k] : W[k*ld + n]
// ---------------------------------------------------------------------------
__global__ void k_pack_bfrag(const float* __restrict__ W, unsigned int* __restrict__ frag,
                             int K, int N, int transposed, int ld) {
  int ktiles = K >> 5, ntiles = N >> 4;
  int total = ktiles * ntiles * 32;
  int t = blockIdx.x * blockDim.x + threadIdx.x;
  if (t >= total) return;
  int lane = t & 31, tile = t >> 5;
  int kt = tile / ntiles, nt = tile % ntiles;
  int half = lane >> 4, n = nt * 16 + (lane & 15);
  unsigned int* out = frag + (size_t)t * 8;
#pragma unroll
  for (int v = 0; v < 8; ++v) {
    int k = kt * 32 + half * 16 + 2 * v;
    float f0 = transposed ? W[(size_t)n * ld + k]     : W[(size_t)k * ld + n];
    float f1 = transposed ? W[(size_t)n * ld + k + 1] : W[(size_t)(k + 1) * ld + n];
    out[v] = (unsigned int)f2bf(f0) | ((unsigned int)f2bf(f1) << 16);
  }
}

__global__ void k_f2bf(const float* __restrict__ s, unsigned short* __restrict__ d, int n) {
  int i = blockIdx.x * blockDim.x + threadIdx.x;
  if (i < n) d[i] = f2bf(s[i]);
}

__global__ void k_fill0(float* __restrict__ p, int n) {
  int i = blockIdx.x * blockDim.x + threadIdx.x;
  if (i < n) p[i] = 0.0f;
}

// ---------------------------------------------------------------------------
// Fragment loaders (16-bit A 16x32 / B 32x16, contiguous k = 16*half + s).
// ---------------------------------------------------------------------------
__device__ __forceinline__ void load_afrag(const unsigned int* __restrict__ rowp,
                                           int kt, int half, FragU& a) {
  int base = kt * 16 + half * 8;
#pragma unroll
  for (int v = 0; v < 4; ++v) a.u[v] = rowp[base + v];
#pragma unroll
  for (int v = 0; v < 4; ++v) a.u[4 + v] = rowp[base + 4 + v];
}

__device__ __forceinline__ void load_bfrag(const unsigned int* __restrict__ Bfrag,
                                           int tile, int lane, FragU& b) {
  const unsigned int* bp = Bfrag + ((size_t)tile * 32 + lane) * 8;
#pragma unroll
  for (int v = 0; v < 8; ++v) b.u[v] = bp[v];
}

__device__ __forceinline__ void load_bfrag_lds(const unsigned int* bs,
                                               int tile, int lane, FragU& b) {
  const unsigned int* bp = bs + ((unsigned)tile * 32 + lane) * 8;
#pragma unroll
  for (int v = 0; v < 8; ++v) b.u[v] = bp[v];
}

// ---------------------------------------------------------------------------
// Streaming GEMM, multi-accumulator + LDS-staged B panel + software pipeline.
// One wave owns one 16-row tile and all CT col tiles; A fragment loaded once
// per k-step and reused for CT WMMAs (A traffic cut CT-fold). The entire B
// panel (KT*CT tiles = 16/32KB) is cooperatively staged in LDS once per
// block: 8 waves share it, eliminating ~200MB of redundant L2 reads per GEMM
// pass, and the ping-pong fragment buffers then wait only on low-latency DS
// loads instead of L2.
// ---------------------------------------------------------------------------
template <int CT, int KT>
__global__ void k_wmma_gemm(const unsigned short* __restrict__ Abf,
                            const unsigned int* __restrict__ Bfrag,
                            float* __restrict__ Out,
                            unsigned short* __restrict__ OutBf,
                            int rows) {
  const int Nout = CT * 16;
  const int K = KT * 32;
  __shared__ unsigned int bs[KT * CT * 256];       // KT*CT tiles * 32 lanes * 8 dw

  // cooperative stage of B panel (all threads; barrier before any early-out)
  const int totalD = KT * CT * 256;
  for (int i = threadIdx.x * 4; i < totalD; i += 256 * 4)
    *(uint4*)&bs[i] = *(const uint4*)&Bfrag[i];
  __syncthreads();

  int wave = threadIdx.x >> 5, lane = threadIdx.x & 31;
  int rowTile = blockIdx.x * (blockDim.x >> 5) + wave;
  if (rowTile >= (rows >> 4)) return;              // wave-uniform
  int half = lane >> 4, l15 = lane & 15;
  const unsigned int* rowp =
      (const unsigned int*)Abf + (size_t)(rowTile * 16 + l15) * (K >> 1);
  v8f acc[CT];
#pragma unroll
  for (int c = 0; c < CT; ++c)
#pragma unroll
    for (int j = 0; j < 8; ++j) acc[c][j] = 0.0f;

  FragU a[2], b[2];
  load_afrag(rowp, 0, half, a[0]);
  load_bfrag_lds(bs, 0, lane, b[0]);
#pragma unroll
  for (int kt = 0; kt < KT; ++kt) {
    if (kt + 1 < KT) load_afrag(rowp, kt + 1, half, a[(kt + 1) & 1]);
#pragma unroll
    for (int ct = 0; ct < CT; ++ct) {
      const int t = kt * CT + ct;
      if (t + 1 < KT * CT) load_bfrag_lds(bs, t + 1, lane, b[(t + 1) & 1]);
      acc[ct] = __builtin_amdgcn_wmma_f32_16x16x32_bf16(
          false, a[kt & 1].v, false, b[t & 1].v, (short)0, acc[ct], false, false);
    }
  }

  int row0 = rowTile * 16 + half * 8;
#pragma unroll
  for (int ct = 0; ct < CT; ++ct) {
    int col = ct * 16 + l15;
#pragma unroll
    for (int j = 0; j < 8; ++j) {
      float f = acc[ct][j];
      Out[(size_t)(row0 + j) * Nout + col] = f;
      if (OutBf) OutBf[(size_t)(row0 + j) * Nout + col] = f2bf(f);
    }
  }
}

// ---------------------------------------------------------------------------
// Edge scatter-add: agg[dst] += m[src]. 32 threads/edge, float4 gather,
// 4x global_atomic_add_f32 (m + agg fit in 192MB L2 -> atomics L2-resident).
// ---------------------------------------------------------------------------
__global__ void k_scatter(const float* __restrict__ m, const long long* __restrict__ ei,
                          float* __restrict__ agg, int nEdges) {
  unsigned int gid = blockIdx.x * blockDim.x + threadIdx.x;
  int e = gid >> 5;
  if (e >= nEdges) return;
  int c = (gid & 31) * 4;
  long long s = ei[e];
  long long d = ei[(size_t)nEdges + e];
  float4 v = *(const float4*)(m + (size_t)s * CH + c);
  float* ap = agg + (size_t)d * CH + c;
  atomicAdd(ap + 0, v.x);
  atomicAdd(ap + 1, v.y);
  atomicAdd(ap + 2, v.z);
  atomicAdd(ap + 3, v.w);
}

// ---------------------------------------------------------------------------
// Fused GRU step: one block per 16-node tile. Waves 0-3 compute gi (A=agg,
// 6 col tiles each), waves 4-7 compute gh (A=h). A fragment loaded once per
// k-step, reused across 6 accumulators; B fragments ping-pong double-buffered
// (software pipeline, fully unrolled; wih/whh panels are 192KB total -> kept
// in global, L0-resident across blocks). Results staged in LDS, then 256
// threads evaluate the GRU gates and rewrite h (f32 + bf16). gi/gh never
// touch global memory (~307MB/step saved).
// ---------------------------------------------------------------------------
__global__ void k_gru_fused(const unsigned short* __restrict__ aggBf,
                            const unsigned short* __restrict__ hBfIn,
                            const unsigned int* __restrict__ wihFrag,
                            const unsigned int* __restrict__ whhFrag,
                            const float* __restrict__ b_ih,
                            const float* __restrict__ b_hh,
                            float* __restrict__ hF,
                            unsigned short* __restrict__ hBf) {
  __shared__ float gi_s[16 * CH3];
  __shared__ float gh_s[16 * CH3];
  int wave = threadIdx.x >> 5, lane = threadIdx.x & 31;
  int half = lane >> 4, l15 = lane & 15;
  int rowTile = blockIdx.x;

  bool isGh = wave >= 4;                           // wave-uniform
  int ntBase = (wave & 3) * 6;
  const unsigned short* A = isGh ? hBfIn : aggBf;
  const unsigned int*  Bf = isGh ? whhFrag : wihFrag;
  float* dst = isGh ? gh_s : gi_s;
  const unsigned int* rowp =
      (const unsigned int*)A + (size_t)(rowTile * 16 + l15) * (CH >> 1);

  v8f acc[6];
#pragma unroll
  for (int c = 0; c < 6; ++c)
#pragma unroll
    for (int j = 0; j < 8; ++j) acc[c][j] = 0.0f;

  FragU a[2], b[2];
  load_afrag(rowp, 0, half, a[0]);
  load_bfrag(Bf, ntBase, lane, b[0]);
#pragma unroll
  for (int kt = 0; kt < 4; ++kt) {
    if (kt + 1 < 4) load_afrag(rowp, kt + 1, half, a[(kt + 1) & 1]);
#pragma unroll
    for (int j = 0; j < 6; ++j) {
      const int t = kt * 6 + j;
      if (t + 1 < 24) {
        int nj = (j + 1 < 6) ? j + 1 : 0;
        int nkt = (j + 1 < 6) ? kt : kt + 1;
        load_bfrag(Bf, nkt * 24 + ntBase + nj, lane, b[(t + 1) & 1]);
      }
      acc[j] = __builtin_amdgcn_wmma_f32_16x16x32_bf16(
          false, a[kt & 1].v, false, b[t & 1].v, (short)0, acc[j], false, false);
    }
  }
#pragma unroll
  for (int j = 0; j < 6; ++j) {
    int col = (ntBase + j) * 16 + l15;
#pragma unroll
    for (int r = 0; r < 8; ++r) dst[(half * 8 + r) * CH3 + col] = acc[j][r];
  }
  __syncthreads();

  int tileBase = rowTile * 16;
#pragma unroll
  for (int i = 0; i < 8; ++i) {
    int idx = threadIdx.x + 256 * i;               // 0..2047 over 16x128
    int nl = idx >> 7, c = idx & 127;
    float gir = gi_s[nl * CH3 + c]           + b_ih[c];
    float giz = gi_s[nl * CH3 + CH + c]      + b_ih[CH + c];
    float gin = gi_s[nl * CH3 + 2 * CH + c]  + b_ih[2 * CH + c];
    float ghr = gh_s[nl * CH3 + c]           + b_hh[c];
    float ghz = gh_s[nl * CH3 + CH + c]      + b_hh[CH + c];
    float ghn = gh_s[nl * CH3 + 2 * CH + c]  + b_hh[2 * CH + c];
    float r = 1.0f / (1.0f + __expf(-(gir + ghr)));
    float z = 1.0f / (1.0f + __expf(-(giz + ghz)));
    float nn = tanhf(gin + r * ghn);
    size_t g = (size_t)(tileBase + nl) * CH + c;
    float hnew = (1.0f - z) * nn + z * hF[g];
    hF[g] = hnew;
    hBf[g] = f2bf(hnew);
  }
}

// ---------------------------------------------------------------------------
// Readout: sum_{n,c} leakyrelu(h[n,c]) * W_pred[c]  -> /N + b_pred
// ---------------------------------------------------------------------------
__global__ void k_readout_partial(const float* __restrict__ h,
                                  const float* __restrict__ wpred,
                                  float* __restrict__ part) {
  __shared__ float red[256];
  float s = 0.0f;
  const int total = N_NODES * CH;
  for (int idx = blockIdx.x * 256 + threadIdx.x; idx < total; idx += gridDim.x * 256) {
    int c = idx & (CH - 1);
    float v = h[idx];
    v = v >= 0.0f ? v : NEG_SLOPE * v;
    s += v * wpred[c];
  }
  red[threadIdx.x] = s;
  __syncthreads();
  for (int off = 128; off; off >>= 1) {
    if (threadIdx.x < off) red[threadIdx.x] += red[threadIdx.x + off];
    __syncthreads();
  }
  if (threadIdx.x == 0) part[blockIdx.x] = red[0];
}

__global__ void k_readout_final(const float* __restrict__ part,
                                const float* __restrict__ bpred,
                                float* __restrict__ out) {
  __shared__ float red[256];
  float s = part[threadIdx.x] + part[threadIdx.x + 256] +
            part[threadIdx.x + 512] + part[threadIdx.x + 768];
  red[threadIdx.x] = s;
  __syncthreads();
  for (int off = 128; off; off >>= 1) {
    if (threadIdx.x < off) red[threadIdx.x] += red[threadIdx.x + off];
    __syncthreads();
  }
  if (threadIdx.x == 0) out[0] = red[0] / (float)N_NODES + bpred[0];
}

// ---------------------------------------------------------------------------
extern "C" void kernel_launch(void* const* d_in, const int* in_sizes, int n_in,
                              void* d_out, int out_size, void* d_ws, size_t ws_size,
                              hipStream_t stream) {
  const float*     x      = (const float*)d_in[0];
  const long long* ei     = (const long long*)d_in[1];  // int64 per reference
  const float*     W_in   = (const float*)d_in[2];
  const float*     W_mpnn = (const float*)d_in[3];
  const float*     W_ih   = (const float*)d_in[4];
  const float*     W_hh   = (const float*)d_in[5];
  const float*     b_ih   = (const float*)d_in[6];
  const float*     b_hh   = (const float*)d_in[7];
  const float*     W_pred = (const float*)d_in[8];
  const float*     b_pred = (const float*)d_in[9];
  float* out = (float*)d_out;

  // workspace layout (bytes)
  char* ws = (char*)d_ws;
  const size_t OFF_H     = 0;                        // 100000*128*4 = 51.2MB
  const size_t OFF_HBF   = 51200000;                 // 25.6MB
  const size_t OFF_M     = 76800000;                 // 51.2MB (xBf aliased here)
  const size_t OFF_AGG   = 128000000;                // 51.2MB
  const size_t OFF_AGGBF = 179200000;                // 25.6MB
  const size_t OFF_WIN   = 204800000;                // 16KB
  const size_t OFF_MPNN  = 204816384;                // 128KB (4 steps x 32KB)
  const size_t OFF_WIH   = 204947456;                // 96KB
  const size_t OFF_WHH   = 205045760;                // 96KB
  const size_t OFF_PART  = 205144064;                // 4KB

  float*          hF     = (float*)(ws + OFF_H);
  unsigned short* hBf    = (unsigned short*)(ws + OFF_HBF);
  float*          mF     = (float*)(ws + OFF_M);
  unsigned short* xBf    = (unsigned short*)(ws + OFF_M);   // dead after embed
  float*          aggF   = (float*)(ws + OFF_AGG);
  unsigned short* aggBf  = (unsigned short*)(ws + OFF_AGGBF);
  unsigned int*   fWin   = (unsigned int*)(ws + OFF_WIN);
  unsigned int*   fMpnn  = (unsigned int*)(ws + OFF_MPNN);
  unsigned int*   fWih   = (unsigned int*)(ws + OFF_WIH);
  unsigned int*   fWhh   = (unsigned int*)(ws + OFF_WHH);
  float*          part   = (float*)(ws + OFF_PART);

  // ---- pack weights into B-fragment layout (bf16) ----
  // W_in [C, F_IN], B(k,n)=W_in[n][k] : K=64, N=128, transposed, ld=64
  k_pack_bfrag<<<2, 256, 0, stream>>>(W_in, fWin, F_IN, CH, 1, F_IN);
  for (int i = 0; i < STEPS; ++i)      // W_mpnn[i] [C, C], B = W : ld=128
    k_pack_bfrag<<<4, 256, 0, stream>>>(W_mpnn + (size_t)i * CH * CH,
                                        fMpnn + (size_t)i * 8192, CH, CH, 0, CH);
  // W_ih/W_hh [3C, C], B(k,n)=W[n][k] : K=128, N=384, transposed, ld=128
  k_pack_bfrag<<<12, 256, 0, stream>>>(W_ih, fWih, CH, CH3, 1, CH);
  k_pack_bfrag<<<12, 256, 0, stream>>>(W_hh, fWhh, CH, CH3, 1, CH);

  // ---- input embedding: h = x @ W_in^T ----
  const int nXE = N_NODES * F_IN;
  k_f2bf<<<(nXE + 255) / 256, 256, 0, stream>>>(x, xBf, nXE);
  {
    int rowTiles = N_NODES / 16;                   // 6250 waves, 8 waves/block
    k_wmma_gemm<8, 2><<<(rowTiles + 7) / 8, 256, 0, stream>>>(xBf, fWin, hF, hBf,
                                                              N_NODES);
  }

  const int nHE = N_NODES * CH;                    // 12.8M
  for (int step = 0; step < STEPS; ++step) {
    k_fill0<<<(nHE + 255) / 256, 256, 0, stream>>>(aggF, nHE);
    {
      int rowTiles = N_NODES / 16;
      k_wmma_gemm<8, 4><<<(rowTiles + 7) / 8, 256, 0, stream>>>(hBf,
          fMpnn + (size_t)step * 8192, mF, nullptr, N_NODES);
    }
    {
      unsigned int thr = (unsigned int)N_EDGES * 32u;
      k_scatter<<<(thr + 255) / 256, 256, 0, stream>>>(mF, ei, aggF, N_EDGES);
    }
    k_f2bf<<<(nHE + 255) / 256, 256, 0, stream>>>(aggF, aggBf, nHE);
    k_gru_fused<<<N_NODES / 16, 256, 0, stream>>>(aggBf, hBf, fWih, fWhh,
                                                  b_ih, b_hh, hF, hBf);
  }

  k_readout_partial<<<1024, 256, 0, stream>>>(hF, W_pred, part);
  k_readout_final<<<1, 256, 0, stream>>>(part, b_pred, out);
}
